// PoseGuidedConditionEncoder_57896159150183
// MI455X (gfx1250) — matrix operand order repaired and checked by reading, not compiled, not run on hardware
//
#include <hip/hip_runtime.h>
#include <hip/hip_bf16.h>

// ---------------------------------------------------------------------------
// PoseGuidedConditionEncoder for MI455X (gfx1250).
// All dense contractions run on v_wmma_f32_16x16x32_bf16 (bf16 A/B, f32 acc).
// ~350 GFLOP of GEMM vs ~hundreds of MB of traffic -> compute bound, so the
// matrix pipe does the work; staging converts f32->bf16 into LDS as packed
// dwords (v_cvt_pk_bf16_f32 + ds_store_b32); fragments come back as 16B-
// aligned contiguous chunks -> ds_load_b128.
// Elementwise / softmax / LN / conv stages are HBM-bound streaming kernels.
// ---------------------------------------------------------------------------

#define B_      2
#define T_      16
#define NTOK_   1024
#define CTOK_   256
#define QDIM_   320
#define INNER_  512
#define CDIM_   1024
#define C_      2
#define FFI_    1280          // FF_INNER
#define SCALE_  0.125f        // 64^-0.5

#define PITCH_  20            // LDS row pitch in dwords: 16B-aligned rows,
                              // 20*r mod 64 distinct for r=0..15 -> no bank conflicts

typedef __attribute__((ext_vector_type(16))) __bf16 v16bf;
typedef __attribute__((ext_vector_type(8)))  float  v8f;

#if __has_builtin(__builtin_amdgcn_cvt_pk_bf16_f32)
typedef __attribute__((ext_vector_type(2))) __bf16 v2bf;
__device__ __forceinline__ unsigned pack2bf(float lo, float hi) {
  return __builtin_bit_cast(unsigned, __builtin_amdgcn_cvt_pk_bf16_f32(lo, hi));
}
#else
__device__ __forceinline__ unsigned bfbits(float x) {
  unsigned u = __builtin_bit_cast(unsigned, x);
  return (u + 0x7FFFu + ((u >> 16) & 1u)) >> 16;   // round-to-nearest-even
}
__device__ __forceinline__ unsigned pack2bf(float lo, float hi) {
  return (bfbits(lo) & 0xFFFFu) | (bfbits(hi) << 16);
}
#endif

// ---------------------------------------------------------------------------
// Batched WMMA GEMM:  C[z] (CMODE) = alpha * A[z] * op(B[z]) + bias
//   block tile 128x64, 8 waves (4x2), each wave 32x32 (2x2 WMMA subtiles)
//   K stepped by 32 through LDS; A staged [m][k], B staged [n][k], as packed
//   bf16 pairs, pitch 20 dwords (16B-aligned rows, conflict-free).
//   Requires M%128==0, N%64==0, K%32==0 (true for every GEMM in this model).
//   CMODE: 0 = store, 1 = add into C (residual), 2 = max with C (frame max)
// ---------------------------------------------------------------------------
template <int TRANSB, int CMODE>
__global__ __launch_bounds__(256) void wmma_gemm_kernel(
    const float* __restrict__ A, const float* __restrict__ B,
    float* __restrict__ C, const float* __restrict__ bias,
    int K, int lda, int ldb, int ldc,
    long long strideA, long long strideB, int divB, long long strideC,
    float alpha)
{
  const int z = blockIdx.z;
  A += (long long)z * strideA;
  B += (long long)(z / divB) * strideB;
  C += (long long)z * strideC;

  const int bm = blockIdx.y * 128;
  const int bn = blockIdx.x * 64;

  __shared__ unsigned As32[128 * PITCH_];   // [m][kpair]
  __shared__ unsigned Bs32[64 * PITCH_];    // [n][kpair]

  const int tid  = threadIdx.x;
  const int wave = tid >> 5;
  const int lane = tid & 31;
  const int wm   = wave & 3;                 // 0..3 along M
  const int wn   = wave >> 2;                // 0..1 along N
  const int lr   = lane & 15;
  const int hA   = (lane < 16) ? 0 : 4;      // A k-half select, dword units
  const int hB   = (lane < 16) ? 0 : 4;      // B k-half select, dword units

  v8f acc[2][2] = {};

  for (int kt = 0; kt < K; kt += 32) {
    // ---- stage A tile (128x32 f32 -> packed bf16) : 8 dwords/thread ----
    #pragma unroll
    for (int i = 0; i < 8; ++i) {
      int idx = tid + i * 256;             // 0..2047
      int r = idx >> 4, p = idx & 15;
      const float* ap = A + (long long)(bm + r) * lda + kt + 2 * p;
      As32[r * PITCH_ + p] = pack2bf(ap[0], ap[1]);
    }
    // ---- stage B tile into [n][k] : 4 dwords/thread ----
    #pragma unroll
    for (int i = 0; i < 4; ++i) {
      int idx = tid + i * 256;             // 0..1023
      int n = idx & 63, p = idx >> 6;
      float x0, x1;
      if (TRANSB) {
        const float* bp = B + (long long)(bn + n) * ldb + kt + 2 * p;
        x0 = bp[0]; x1 = bp[1];
      } else {
        x0 = B[(long long)(kt + 2 * p)     * ldb + bn + n];
        x1 = B[(long long)(kt + 2 * p + 1) * ldb + bn + n];
      }
      Bs32[n * PITCH_ + p] = pack2bf(x0, x1);
    }
    if (kt + 32 < K) {  // global_prefetch_b8 of the next K tile
      __builtin_prefetch(A + (long long)(bm + (tid & 127)) * lda + kt + 32, 0, 1);
      __builtin_prefetch(TRANSB ? (B + (long long)(bn + (tid & 63)) * ldb + kt + 32)
                                : (B + (long long)(kt + 32 + (tid & 31)) * ldb + bn),
                         0, 1);
    }
    __syncthreads();

    // ---- per-wave fragments (ISA 7.12.2 wave32 layouts) ----
    v16bf afrag[2], bfrag[2];
    #pragma unroll
    for (int i = 0; i < 2; ++i) {
      int base = (wm * 32 + i * 16 + lr) * PITCH_ + hA;
      unsigned au[8];
      #pragma unroll
      for (int u = 0; u < 4; ++u) {        // k 0-7 / 8-15   (16B aligned chunk)
        au[u]     = As32[base + u];
        au[u + 4] = As32[base + 8 + u];    // k 16-23 / 24-31 (16B aligned chunk)
      }
      afrag[i] = __builtin_bit_cast(v16bf, au);
    }
    #pragma unroll
    for (int j = 0; j < 2; ++j) {
      int base = (wn * 32 + j * 16 + lr) * PITCH_ + hB;
      unsigned bu[8];
      #pragma unroll
      for (int u = 0; u < 4; ++u) {        // k 0-7 / 8-15, k 16-23 / 24-31 (16B aligned chunks)
        bu[u]     = Bs32[base + u];
        bu[u + 4] = Bs32[base + 8 + u];
      }
      bfrag[j] = __builtin_bit_cast(v16bf, bu);
    }
    #pragma unroll
    for (int i = 0; i < 2; ++i)
      #pragma unroll
      for (int j = 0; j < 2; ++j)
        acc[i][j] = __builtin_amdgcn_wmma_f32_16x16x32_bf16(
            false, afrag[i], false, bfrag[j], (short)0, acc[i][j], false, false);
    __syncthreads();
  }

  // ---- epilogue: C/D layout = 8 VGPRs, lanes<16 rows v, lanes>=16 rows v+8 ----
  #pragma unroll
  for (int i = 0; i < 2; ++i) {
    #pragma unroll
    for (int j = 0; j < 2; ++j) {
      int col  = bn + wn * 32 + j * 16 + lr;
      float bv = bias ? bias[col] : 0.f;
      #pragma unroll
      for (int v = 0; v < 8; ++v) {
        int row = bm + wm * 32 + i * 16 + ((lane < 16) ? v : v + 8);
        float val = acc[i][j][v] * alpha + bv;
        long long off = (long long)row * ldc + col;
        if      (CMODE == 1) C[off] += val;
        else if (CMODE == 2) C[off]  = fmaxf(C[off], val);
        else                 C[off]  = val;
      }
    }
  }
}

// ---------------------------------------------------------------------------
// pose 'B D T H W -> B T (H W) D'
__global__ void pose_transpose_kernel(const float* __restrict__ pose,
                                      float* __restrict__ out, int total)
{
  int i = blockIdx.x * blockDim.x + threadIdx.x;
  if (i >= total) return;
  int d = i % QDIM_;
  int n = (i / QDIM_) % NTOK_;
  int t = (i / (QDIM_ * NTOK_)) % T_;
  int b = i / (QDIM_ * NTOK_ * T_);
  int h = n >> 5, w = n & 31;
  out[i] = pose[(((long long)(b * QDIM_ + d) * T_ + t) * 32 + h) * 32 + w];
}

// LayerNorm over last dim = 320
__global__ __launch_bounds__(64) void layernorm_kernel(
    const float* __restrict__ x, float* __restrict__ y,
    const float* __restrict__ w, const float* __restrict__ b)
{
  long long row = blockIdx.x;
  const float* xr = x + row * QDIM_;
  float*       yr = y + row * QDIM_;
  __shared__ float red[64];
  int tid = threadIdx.x;
  float s = 0.f;
  for (int i = tid; i < QDIM_; i += 64) s += xr[i];
  red[tid] = s; __syncthreads();
  for (int off = 32; off > 0; off >>= 1) {
    if (tid < off) red[tid] += red[tid + off];
    __syncthreads();
  }
  float mu = red[0] * (1.f / QDIM_); __syncthreads();
  float vv = 0.f;
  for (int i = tid; i < QDIM_; i += 64) { float d = xr[i] - mu; vv += d * d; }
  red[tid] = vv; __syncthreads();
  for (int off = 32; off > 0; off >>= 1) {
    if (tid < off) red[tid] += red[tid + off];
    __syncthreads();
  }
  float rstd = rsqrtf(red[0] * (1.f / QDIM_) + 1e-5f);
  for (int i = tid; i < QDIM_; i += 64) yr[i] = (xr[i] - mu) * rstd * w[i] + b[i];
}

// softmax over the agent-kv attention, doing the torch-faithful reshuffle:
//   out[((c*B+b)*T+t)*M+m, n] = softmax_n( flat[b,t,m, n*C+c] + agent_bias[m,n] )
__global__ __launch_bounds__(256) void attn_softmax_kernel(
    const float* __restrict__ flat, const float* __restrict__ bias,
    float* __restrict__ out)
{
  long long r = blockIdx.x;                 // (c,b,t,m)
  int m = (int)(r % NTOK_);
  long long cbt = r / NTOK_;
  int t = (int)(cbt % T_);
  long long cb = cbt / T_;
  int b = (int)(cb % B_);
  int c = (int)(cb / B_);
  const float* src = flat + (((long long)(b * T_ + t) * NTOK_ + m) * (C_ * CTOK_));
  int n = threadIdx.x;
  float v = src[n * C_ + c] + bias[(long long)m * CTOK_ + n];
  __shared__ float red[256];
  red[n] = v; __syncthreads();
  for (int off = 128; off > 0; off >>= 1) {
    if (n < off) red[n] = fmaxf(red[n], red[n + off]);
    __syncthreads();
  }
  float mx = red[0]; __syncthreads();
  float e = __expf(v - mx);
  red[n] = e; __syncthreads();
  for (int off = 128; off > 0; off >>= 1) {
    if (n < off) red[n] += red[n + off];
    __syncthreads();
  }
  out[r * CTOK_ + n] = e / red[0];
}

// softmax over width 1024 (query->agent), bias = query_bias[row%NTOK, :]
__global__ __launch_bounds__(256) void aq_softmax_kernel(
    const float* __restrict__ sin, const float* __restrict__ qbias,
    float* __restrict__ out)
{
  long long row = blockIdx.x;
  int n = (int)(row % NTOK_);
  const float* src = sin + row * NTOK_;
  int tid = threadIdx.x;
  float v[4]; float mx = -3.4e38f;
  #pragma unroll
  for (int i = 0; i < 4; ++i) {
    int col = tid + i * 256;
    v[i] = src[col] + qbias[(long long)n * NTOK_ + col];
    mx = fmaxf(mx, v[i]);
  }
  __shared__ float red[256];
  red[tid] = mx; __syncthreads();
  for (int off = 128; off > 0; off >>= 1) {
    if (tid < off) red[tid] = fmaxf(red[tid], red[tid + off]);
    __syncthreads();
  }
  mx = red[0]; __syncthreads();
  float s = 0.f;
  #pragma unroll
  for (int i = 0; i < 4; ++i) { v[i] = __expf(v[i] - mx); s += v[i]; }
  red[tid] = s; __syncthreads();
  for (int off = 128; off > 0; off >>= 1) {
    if (tid < off) red[tid] += red[tid + off];
    __syncthreads();
  }
  float inv = 1.f / red[0];
  #pragma unroll
  for (int i = 0; i < 4; ++i) out[row * NTOK_ + tid + i * 256] = v[i] * inv;
}

// depthwise 3x3 'SAME' over v reshaped to [B*C, INNER, 16,16], + bias, max over C
__global__ void dwconv_max_kernel(const float* __restrict__ vws,
                                  const float* __restrict__ w,
                                  const float* __restrict__ bias,
                                  float* __restrict__ out)
{
  int i = blockIdx.x * blockDim.x + threadIdx.x;
  if (i >= B_ * INNER_ * CTOK_) return;
  int pix = i % CTOK_;
  int e   = (i / CTOK_) % INNER_;
  int b   = i / (CTOK_ * INNER_);
  int h = pix >> 4, wc = pix & 15;
  const float* wk = w + e * 9;
  float best = -3.4e38f;
  for (int c = 0; c < C_; ++c) {
    float acc = bias[e];
    #pragma unroll
    for (int dy = -1; dy <= 1; ++dy)
      #pragma unroll
      for (int dx = -1; dx <= 1; ++dx) {
        int hh = h + dy, ww = wc + dx;
        if (hh >= 0 && hh < 16 && ww >= 0 && ww < 16)
          acc += wk[(dy + 1) * 3 + (dx + 1)] *
                 vws[((long long)((b * C_ + c) * CTOK_ + hh * 16 + ww)) * INNER_ + e];
      }
    best = fmaxf(best, acc);
  }
  out[i] = best;   // [B, INNER, 16,16]
}

// bilinear x2 upsample (half-pixel) 16x16 -> 32x32, write v_tok[b,n,e]
__global__ void bilinear_vtok_kernel(const float* __restrict__ vd,
                                     float* __restrict__ vtok)
{
  int i = blockIdx.x * blockDim.x + threadIdx.x;
  if (i >= B_ * NTOK_ * INNER_) return;
  int e = i % INNER_;
  int n = (i / INNER_) % NTOK_;
  int b = i / (INNER_ * NTOK_);
  int h = n >> 5, w = n & 31;
  float sy = (h + 0.5f) * 0.5f - 0.5f;
  float sx = (w + 0.5f) * 0.5f - 0.5f;
  int y0 = (int)floorf(sy), x0 = (int)floorf(sx);
  float fy = sy - y0, fx = sx - x0;
  int y1 = min(15, max(0, y0 + 1)); y0 = min(15, max(0, y0));
  int x1 = min(15, max(0, x0 + 1)); x0 = min(15, max(0, x0));
  const float* base = vd + ((long long)b * INNER_ + e) * CTOK_;
  float v00 = base[y0 * 16 + x0], v01 = base[y0 * 16 + x1];
  float v10 = base[y1 * 16 + x0], v11 = base[y1 * 16 + x1];
  vtok[i] = (1.f - fy) * ((1.f - fx) * v00 + fx * v01) +
            fy * ((1.f - fx) * v10 + fx * v11);
}

// x[b,t,n,:] += v_tok[b,n,:]
__global__ void add_vtok_kernel(float* __restrict__ x,
                                const float* __restrict__ vtok)
{
  int i = blockIdx.x * blockDim.x + threadIdx.x;
  if (i >= B_ * T_ * NTOK_ * INNER_) return;
  int e = i % INNER_;
  int n = (i / INNER_) % NTOK_;
  int b = i / (INNER_ * NTOK_ * T_);
  x[i] += vtok[((long long)b * NTOK_ + n) * INNER_ + e];
}

// GEGLU: out[r,j] = p[r,j] * gelu(p[r, FFI+j]), exact gelu
__global__ void geglu_kernel(const float* __restrict__ p,
                             float* __restrict__ out, long long rows)
{
  long long i = (long long)blockIdx.x * blockDim.x + threadIdx.x;
  if (i >= rows * FFI_) return;
  long long r = i / FFI_;
  int j = (int)(i % FFI_);
  float a = p[r * (2 * FFI_) + j];
  float g = p[r * (2 * FFI_) + FFI_ + j];
  out[i] = a * (0.5f * g * (1.f + erff(g * 0.70710678118654752f)));
}

// ---------------------------------------------------------------------------
static inline void gemm(hipStream_t s, const float* A, const float* B, float* C,
                        const float* bias, int M, int N, int K,
                        int lda, int ldb, int ldc,
                        long long sA, long long sB, int divB, long long sC,
                        int Z, int transB, float alpha, int cmode)
{
  dim3 g(N / 64, M / 128, Z), blk(256);
#define LNCH(TB, CM) wmma_gemm_kernel<TB, CM><<<g, blk, 0, s>>>( \
      A, B, C, bias, K, lda, ldb, ldc, sA, sB, divB, sC, alpha)
  if (transB) {
    if (cmode == 0) LNCH(1, 0); else if (cmode == 1) LNCH(1, 1); else LNCH(1, 2);
  } else {
    if (cmode == 0) LNCH(0, 0); else if (cmode == 1) LNCH(0, 1); else LNCH(0, 2);
  }
#undef LNCH
}

extern "C" void kernel_launch(void* const* d_in, const int* in_sizes, int n_in,
                              void* d_out, int out_size, void* d_ws, size_t ws_size,
                              hipStream_t stream)
{
  (void)in_sizes; (void)n_in; (void)out_size; (void)ws_size;

  const float* in_hs    = (const float*)d_in[0];
  const float* in_ctx   = (const float*)d_in[1];
  const float* in_pose  = (const float*)d_in[2];
  const float* in_Wq    = (const float*)d_in[3];
  const float* in_Wk    = (const float*)d_in[4];
  const float* in_Wv    = (const float*)d_in[5];
  const float* in_Wa    = (const float*)d_in[6];
  const float* in_ab    = (const float*)d_in[7];
  const float* in_qb    = (const float*)d_in[8];
  const float* in_dwcw  = (const float*)d_in[9];
  const float* in_dwcb  = (const float*)d_in[10];
  const float* in_Wo    = (const float*)d_in[11];
  const float* in_nw    = (const float*)d_in[12];
  const float* in_nb    = (const float*)d_in[13];
  const float* in_fnw   = (const float*)d_in[14];
  const float* in_fnb   = (const float*)d_in[15];
  const float* in_fw1   = (const float*)d_in[16];
  const float* in_fb1   = (const float*)d_in[17];
  const float* in_fw2   = (const float*)d_in[18];
  const float* in_fb2   = (const float*)d_in[19];

  // ---- workspace arena (floats) ----
  const long long HS  = (long long)B_ * T_ * NTOK_ * QDIM_;       // 10,485,760
  const long long BIG = (long long)B_ * T_ * NTOK_ * INNER_;      // 16,777,216

  float* ws   = (float*)d_ws;
  float* hs   = ws;                                   // HS
  float* pt   = hs   + HS;                            // HS (pose_t)
  float* kws  = pt   + HS;                            // [1024,512]
  float* vws  = kws  + (long long)B_ * C_ * CTOK_ * INNER_;
  float* conv = vws  + (long long)B_ * C_ * CTOK_ * INNER_;   // B*INNER*256
  float* vtok = conv + (long long)B_ * INNER_ * CTOK_;        // B*NTOK*INNER
  float* nh   = vtok + (long long)B_ * NTOK_ * INNER_;        // HS (LN output)
  float* tail = nh   + HS;
  // tail reused: block loop -> atok|attnf(q)|attnr(x)|akv|aq ; FF -> p|gg
  float* atok  = tail;
  float* attnf = atok  + BIG;   // later reused as q
  float* attnr = attnf + BIG;   // later reused as x
  float* akv   = attnr + BIG;   // agent_kv
  float* aq    = akv   + BIG;   // 33,554,432
  float* pbuf  = tail;                          // [32768, 2560]
  float* ggbuf = tail + (long long)B_ * T_ * NTOK_ * 2 * FFI_;

  // hs <- hidden_state ((B T) N H D is already contiguous as B,T,N,QDIM)
  hipMemcpyAsync(hs, in_hs, HS * sizeof(float), hipMemcpyDeviceToDevice, stream);

  // pose transpose
  {
    int total = (int)HS;
    pose_transpose_kernel<<<(total + 255) / 256, 256, 0, stream>>>(in_pose, pt, total);
  }

  for (int blk = 0; blk < 2; ++blk) {
    const float* Wq = in_Wq + (long long)blk * QDIM_ * INNER_;
    const float* Wk = in_Wk + (long long)blk * CDIM_ * INNER_;
    const float* Wv = in_Wv + (long long)blk * CDIM_ * INNER_;
    const float* Wa = in_Wa + (long long)blk * QDIM_ * INNER_;
    const float* Wo = in_Wo + (long long)blk * INNER_ * QDIM_;
    const float* ab = in_ab + (long long)blk * NTOK_ * CTOK_;
    const float* qb = in_qb + (long long)blk * NTOK_ * NTOK_;
    const float* dw = in_dwcw + (long long)blk * INNER_ * 9;
    const float* db = in_dwcb + (long long)blk * INNER_;

    // k = context @ Wk ; v = context @ Wv   ([1024,1024] x [1024,512])
    gemm(stream, in_ctx, Wk, kws, nullptr, B_ * C_ * CTOK_, INNER_, CDIM_,
         CDIM_, INNER_, INNER_, 0, 0, 1, 0, 1, 0, 1.f, 0);
    gemm(stream, in_ctx, Wv, vws, nullptr, B_ * C_ * CTOK_, INNER_, CDIM_,
         CDIM_, INNER_, INNER_, 0, 0, 1, 0, 1, 0, 1.f, 0);

    // a_tok = pose_t @ Wa   ([32768,320] x [320,512])
    gemm(stream, pt, Wa, atok, nullptr, B_ * T_ * NTOK_, INNER_, QDIM_,
         QDIM_, INNER_, INNER_, 0, 0, 1, 0, 1, 0, 1.f, 0);

    // attn_flat = SCALE * a_tok @ kf^T   (z over B*T, kf batched per b)
    gemm(stream, atok, kws, attnf, nullptr, NTOK_, C_ * CTOK_, INNER_,
         INNER_, INNER_, C_ * CTOK_,
         (long long)NTOK_ * INNER_, (long long)C_ * CTOK_ * INNER_, T_,
         (long long)NTOK_ * (C_ * CTOK_), B_ * T_, 1, SCALE_, 0);

    // softmax (+agent bias, torch-faithful reshuffle) -> attnr [C,B,T,M,CTOK]
    attn_softmax_kernel<<<C_ * B_ * T_ * NTOK_, 256, 0, stream>>>(attnf, ab, attnr);

    // agent_kv = max_c (attn_r @ v)
    for (int c = 0; c < C_; ++c)
      gemm(stream, attnr + (long long)c * B_ * T_ * NTOK_ * CTOK_,
           vws + (long long)c * CTOK_ * INNER_, akv, nullptr,
           NTOK_, INNER_, CTOK_, CTOK_, INNER_, INNER_,
           (long long)NTOK_ * CTOK_, (long long)C_ * CTOK_ * INNER_, T_,
           (long long)NTOK_ * INNER_, B_ * T_, 0, 1.f, (c == 0) ? 0 : 2);

    // depthwise conv + frame max -> bilinear x2 -> v_tok
    dwconv_max_kernel<<<(B_ * INNER_ * CTOK_ + 255) / 256, 256, 0, stream>>>(
        vws, dw, db, conv);
    bilinear_vtok_kernel<<<(B_ * NTOK_ * INNER_ + 255) / 256, 256, 0, stream>>>(
        conv, vtok);

    // nh = LN(hs)
    layernorm_kernel<<<B_ * T_ * NTOK_, 64, 0, stream>>>(hs, nh, in_nw, in_nb);

    // q = nh @ Wq  (reuse attnf region)
    float* q = attnf;
    gemm(stream, nh, Wq, q, nullptr, B_ * T_ * NTOK_, INNER_, QDIM_,
         QDIM_, INNER_, INNER_, 0, 0, 1, 0, 1, 0, 1.f, 0);

    // aq = SCALE * q @ a_tok^T   (z over B*T)
    gemm(stream, q, atok, aq, nullptr, NTOK_, NTOK_, INNER_,
         INNER_, INNER_, NTOK_,
         (long long)NTOK_ * INNER_, (long long)NTOK_ * INNER_, 1,
         (long long)NTOK_ * NTOK_, B_ * T_, 1, SCALE_, 0);
    aq_softmax_kernel<<<B_ * T_ * NTOK_, 256, 0, stream>>>(aq, qb, aq);

    // x = aq @ agent_kv  (reuse attnr region)
    float* x = attnr;
    gemm(stream, aq, akv, x, nullptr, NTOK_, INNER_, NTOK_,
         NTOK_, INNER_, INNER_,
         (long long)NTOK_ * NTOK_, (long long)NTOK_ * INNER_, 1,
         (long long)NTOK_ * INNER_, B_ * T_, 0, 1.f, 0);

    // x += v_tok (broadcast over T)
    add_vtok_kernel<<<(int)((BIG + 255) / 256), 256, 0, stream>>>(x, vtok);

    // hs += x @ Wo
    gemm(stream, x, Wo, hs, nullptr, B_ * T_ * NTOK_, QDIM_, INNER_,
         INNER_, QDIM_, QDIM_, 0, 0, 1, 0, 1, 0, 1.f, 1);
  }

  // ---- FeedForward: hs += GEGLU(LN(hs)) ----
  layernorm_kernel<<<B_ * T_ * NTOK_, 64, 0, stream>>>(hs, nh, in_fnw, in_fnb);
  gemm(stream, nh, in_fw1, pbuf, in_fb1, B_ * T_ * NTOK_, 2 * FFI_, QDIM_,
       QDIM_, 2 * FFI_, 2 * FFI_, 0, 0, 1, 0, 1, 0, 1.f, 0);
  {
    long long total = (long long)B_ * T_ * NTOK_ * FFI_;
    geglu_kernel<<<(int)((total + 255) / 256), 256, 0, stream>>>(
        pbuf, ggbuf, (long long)B_ * T_ * NTOK_);
  }
  gemm(stream, ggbuf, in_fw2, hs, in_fb2, B_ * T_ * NTOK_, QDIM_, FFI_,
       FFI_, QDIM_, QDIM_, 0, 0, 1, 0, 1, 0, 1.f, 1);

  hipMemcpyAsync(d_out, hs, HS * sizeof(float), hipMemcpyDeviceToDevice, stream);
}
